// MultiHeadAttention_47553877901603
// MI455X (gfx1250) — matrix operand order repaired and checked by reading, not compiled, not run on hardware
//
#include <hip/hip_runtime.h>

// ---------------------------------------------------------------------------
// MultiHeadAttention for MI455X (gfx1250, wave32, WMMA + TDM).
// N=2, S=4096, E=512, H=8, D=64.
//
// Roofline: mandatory HBM traffic is dominated by the 1.07 GB fp32 attention
// tensor the reference returns (~46us at 23.3 TB/s). Matrix math (~120 GFLOP
// incl. the QK^T recompute) is trivial for the bf16 WMMA units, so the design
// minimizes memory traffic: fused two-pass flash-style attention (pass1 =
// online max/sum stats, pass2 = recompute scores, emit final probs once +
// P@V), never spilling raw energy to HBM. K/Q tiles are staged to LDS by the
// Tensor Data Mover (tensor_load_to_lds, double-buffered, s_wait_tensorcnt)
// so DMA overlaps the WMMA loop. All GEMMs use v_wmma_f32_16x16x32_bf16.
// ---------------------------------------------------------------------------

typedef __attribute__((ext_vector_type(16))) __bf16          v16bf;
typedef __attribute__((ext_vector_type(16))) unsigned short  v16us;
typedef __attribute__((ext_vector_type(8)))  unsigned short  us8;
typedef __attribute__((ext_vector_type(8)))  float           v8f;
typedef __attribute__((ext_vector_type(4)))  unsigned int    v4u;
typedef __attribute__((ext_vector_type(8)))  int             v8i;
typedef __attribute__((ext_vector_type(4)))  int             v4i;

__device__ __forceinline__ unsigned short f2bf(float f) {
  unsigned int u = __float_as_uint(f);
  u += 0x7FFFu + ((u >> 16) & 1u);          // round-to-nearest-even
  return (unsigned short)(u >> 16);
}
__device__ __forceinline__ float bf2f(unsigned short h) {
  return __uint_as_float(((unsigned int)h) << 16);
}

__device__ __forceinline__ v8f wmma_bf16(v16us a, v16us b, v8f c) {
  return __builtin_amdgcn_wmma_f32_16x16x32_bf16(
      false, __builtin_bit_cast(v16bf, a),
      false, __builtin_bit_cast(v16bf, b),
      (short)0, c, false, false);
}

// Low 32 bits of a generic pointer to __shared__ == LDS byte offset
// (flat LDS aperture).
__device__ __forceinline__ unsigned lds_off(const void* p) {
  return (unsigned)(size_t)p;
}

#if __has_builtin(__builtin_amdgcn_tensor_load_to_lds)
#define HAVE_TDM 1
#else
#define HAVE_TDM 0
#endif

// TDM: DMA a 64x64 bf16 tile (row stride `stride_elems` in memory) into LDS
// with hardware padding of 16 B per 128 B row -> LDS row stride 72 halves,
// matching the WMMA fragment loaders. Tracked by TENSORcnt.
// This toolchain's builtin is the 6-arg form:
//   (uint32x4 g0, int32x8 g1, int32x4 g2, int32x4 g3, int32x8 pad, i32 cpol)
__device__ __forceinline__ void tdm_load_tile64(unsigned lds_addr,
                                                const void* gaddr,
                                                int stride_elems) {
#if HAVE_TDM
  const unsigned long long ga = (unsigned long long)(size_t)gaddr;
  v4u g0;
  g0[0] = 1u;                                        // count=1 (valid), user mode
  g0[1] = lds_addr;                                  // lds_addr (bytes)
  g0[2] = (unsigned)(ga & 0xffffffffu);              // global_addr[31:0]
  g0[3] = (unsigned)((ga >> 32) & 0x01ffffffu)       // global_addr[56:32]
        | (2u << 30);                                // type=2 ("image")
  v8i g1;
  g1[0] = (int)((1u << 16)      // data_size = 2 bytes
              | (1u << 20)      // pad_enable
              | (4u << 22)      // pad_interval: 32 DWORDs (=128 B = one row)
              | (3u << 25));    // pad_amount:    4 DWORDs (=16 B)
  g1[1] = (int)(64u << 16);     // tensor_dim0[15:0] = 64 (atomic_barrier=0)
  g1[2] = (int)(64u << 16);     // tensor_dim0 hi=0 | tensor_dim1[15:0] = 64
  g1[3] = (int)(64u << 16);     // tensor_dim1 hi=0 | tile_dim0 = 64
  g1[4] = 64;                   // tile_dim1 = 64 | tile_dim2 = 0
  g1[5] = stride_elems;         // tensor_dim0_stride[31:0]
  g1[6] = 0;                    // stride0 hi | tensor_dim1_stride lo
  g1[7] = 0;
  v4i z4 = {0, 0, 0, 0};        // groups 2/3 unused (2D tensor)
  v8i z8 = {0, 0, 0, 0, 0, 0, 0, 0};   // trailing VADDR slot: NULL
  __builtin_amdgcn_tensor_load_to_lds(g0, g1, z4, z4, z8, 0);
#else
  (void)lds_addr; (void)gaddr; (void)stride_elems;
#endif
}

__device__ __forceinline__ void tdm_wait(int pending) {
#if HAVE_TDM
  if (pending) __builtin_amdgcn_s_wait_tensorcnt(1);
  else         __builtin_amdgcn_s_wait_tensorcnt(0);
#else
  (void)pending;
#endif
}

// A fragment 16x32 bf16 from LDS (row-major, stride 72 halves).
// Lanes 0-15: row M=lane, K = k0+0..7 and k0+16..23.
// Lanes 16-31: row M=lane-16, K = k0+8..15 and k0+24..31.
__device__ __forceinline__ v16us load_a(const unsigned short* __restrict__ s,
                                        int m0, int k0, int lane) {
  const int row = m0 + (lane & 15);
  const int kb  = k0 + ((lane & 16) ? 8 : 0);
  const us8* p = (const us8*)(s + row * 72 + kb);
  us8 lo = p[0];
  us8 hi = p[2];                            // +16 halves
  v16us r;
#pragma unroll
  for (int i = 0; i < 8; ++i) { r[i] = lo[i]; r[8 + i] = hi[i]; }
  return r;
}

// B fragment 32x16 bf16 from LDS stored TRANSPOSED as Bt[n][k] (stride 72).
// Lanes 0-15: N=lane, K = k0+0..7 and k0+16..23. Lanes 16-31: N=lane-16, K = k0+8..15 and k0+24..31.
__device__ __forceinline__ v16us load_b(const unsigned short* __restrict__ s,
                                        int n0, int k0, int lane) {
  const int col = n0 + (lane & 15);
  const int kb  = k0 + ((lane & 16) ? 8 : 0);
  const us8* p = (const us8*)(s + col * 72 + kb);
  us8 lo = p[0];
  us8 hi = p[2];
  v16us r;
#pragma unroll
  for (int i = 0; i < 8; ++i) { r[i] = lo[i]; r[8 + i] = hi[i]; }
  return r;
}

// ---------------------------------------------------------------------------
// Kernel 1: convert W [n][k] f32 -> Wt [k][n] bf16 (x @ W.T becomes x @ Wt).
// Order in ws: m=0 -> W_q (pre-scaled by 1/sqrt(512)), 1 -> W_k, 2 -> W_v,
// 3 -> W_o.
// ---------------------------------------------------------------------------
__global__ __launch_bounds__(256) void wt_convert(
    const float* __restrict__ Wv, const float* __restrict__ Wk,
    const float* __restrict__ Wq, const float* __restrict__ Wo,
    unsigned short* __restrict__ dstBase) {
  const int m = blockIdx.y;
  const float* src = (m == 0) ? Wq : (m == 1) ? Wk : (m == 2) ? Wv : Wo;
  unsigned short* dst = dstBase + (size_t)m * 262144;
  const int idx = blockIdx.x * 256 + threadIdx.x;   // k*512 + n
  const int k = idx >> 9, n = idx & 511;
  float x = src[n * 512 + k];
  if (m == 0) x *= 0.04419417382415922f;            // fold 1/sqrt(512) into W_q
  dst[idx] = f2bf(x);
}

// ---------------------------------------------------------------------------
// Kernel 2: Y_bf16[8192,512] = X_f32[8192,512] @ Wt_bf16[512,512]
// 64x64 block tile, 256 threads = 8 waves in 4(M) x 2(N) grid,
// each wave owns a 16x32 C slice (2 accumulator fragments).
// ---------------------------------------------------------------------------
__global__ __launch_bounds__(256) void proj_gemm(
    const float* __restrict__ X, const unsigned short* __restrict__ Wt,
    unsigned short* __restrict__ Y) {
  __shared__ alignas(16) unsigned short As[64 * 72];
  __shared__ alignas(16) unsigned short Bt[64 * 72];   // [n][k]
  const int tid = threadIdx.x, lane = tid & 31, wave = tid >> 5;
  const int wm = wave & 3, wn = wave >> 2;
  const int row0 = blockIdx.x * 64, col0 = blockIdx.y * 64;
  const int lr = tid >> 2, lc = (tid & 3) * 16;
  v8f acc0 = {}, acc1 = {};
  for (int kt = 0; kt < 512; kt += 64) {
    {
      const float* src = X + (size_t)(row0 + lr) * 512 + kt + lc;
      unsigned short* dst = As + lr * 72 + lc;
#pragma unroll
      for (int i = 0; i < 16; ++i) dst[i] = f2bf(src[i]);
    }
    {
      const unsigned short* src = Wt + (size_t)(kt + lr) * 512 + col0 + lc;
#pragma unroll
      for (int i = 0; i < 16; ++i) Bt[(lc + i) * 72 + lr] = src[i];
    }
    __syncthreads();
#pragma unroll
    for (int ks = 0; ks < 64; ks += 32) {
      v16us a  = load_a(As, wm * 16, ks, lane);
      v16us b0 = load_b(Bt, wn * 32, ks, lane);
      v16us b1 = load_b(Bt, wn * 32 + 16, ks, lane);
      acc0 = wmma_bf16(a, b0, acc0);
      acc1 = wmma_bf16(a, b1, acc1);
    }
    __syncthreads();
  }
  const int half8 = (lane >> 4) * 8, l16 = lane & 15;
#pragma unroll
  for (int v = 0; v < 8; ++v) {
    const int r = row0 + wm * 16 + half8 + v;
    const int c = col0 + wn * 32 + l16;
    Y[(size_t)r * 512 + c]      = f2bf(acc0[v]);
    Y[(size_t)r * 512 + c + 16] = f2bf(acc1[v]);
  }
}

// ---------------------------------------------------------------------------
// Kernel 3: fused attention. One workgroup per (batch n, head h, 64-query
// tile). Pass 1 streams K tiles (TDM, double-buffered) computing online row
// max / sum-of-exp. Pass 2 recomputes scores, writes final probabilities
// (fp32 to d_out, bf16 to LDS) and accumulates O += P @ V. Q arrives
// pre-scaled by 1/sqrt(512) (folded into W_q), so it is DMA'd verbatim.
// ---------------------------------------------------------------------------
__global__ __launch_bounds__(256) void attn_kernel(
    const unsigned short* __restrict__ Qp, const unsigned short* __restrict__ Kp,
    const unsigned short* __restrict__ Vp, const int* __restrict__ mask,
    unsigned short* __restrict__ Op, float* __restrict__ att) {
  __shared__ alignas(16) unsigned short Qs[64 * 72];      // [q][d] pre-scaled
  __shared__ alignas(16) unsigned short Ks[2][64 * 72];   // [key][d] dbl-buf
  __shared__ alignas(16) unsigned short Vt[64 * 72];      // [d][key]
  __shared__ alignas(16) unsigned short Ps[64 * 72];      // [q][key] probs
  __shared__ float redbuf[2][64];
  __shared__ float row_m[64];
  __shared__ float row_l[64];
  __shared__ float row_inv[64];

  const int qt = blockIdx.x, h = blockIdx.y, n = blockIdx.z;
  const int tid = threadIdx.x, lane = tid & 31, wave = tid >> 5;
  const int wm = wave & 3, wn = wave >> 2;
  const int half8 = (lane >> 4) * 8, l16 = lane & 15;
  const int lr = tid >> 2, lc = (tid & 3) * 16;
  const float MNEG = -1.0e30f;

  const size_t qrow0 = (size_t)n * 4096 + (size_t)qt * 64;
  const unsigned short* kb0 = Kp + ((size_t)n * 4096) * 512 + h * 64;
  const unsigned short* vb0 = Vp + ((size_t)n * 4096) * 512 + h * 64;
  const int* mrow = mask + (size_t)n * 4096;

  if (tid < 64) { row_m[tid] = -3.0e38f; row_l[tid] = 0.0f; }
  if (wave == 0) {
    tdm_load_tile64(lds_off(Qs), Qp + qrow0 * 512 + h * 64, 512);
    tdm_load_tile64(lds_off(Ks[0]), kb0, 512);
  }

  // ---------------- pass 1: statistics ----------------
  for (int kt = 0; kt < 64; ++kt) {
    if (wave == 0) {
      if (kt + 1 < 64) {
        tdm_load_tile64(lds_off(Ks[(kt + 1) & 1]),
                        kb0 + (size_t)(kt + 1) * 64 * 512, 512);
        tdm_wait(1);                 // tile kt (and Q) complete, kt+1 in flight
      } else {
        tdm_wait(0);
      }
    }
    __syncthreads();
    const unsigned short* Kc = Ks[kt & 1];

    v8f sc0 = {}, sc1 = {};
#pragma unroll
    for (int ks = 0; ks < 64; ks += 32) {
      v16us a  = load_a(Qs, wm * 16, ks, lane);
      v16us b0 = load_b(Kc, wn * 32, ks, lane);
      v16us b1 = load_b(Kc, wn * 32 + 16, ks, lane);
      sc0 = wmma_bf16(a, b0, sc0);
      sc1 = wmma_bf16(a, b1, sc1);
    }
    const int kg = kt * 64 + wn * 32 + l16;
    const bool mk0 = (mrow[kg] == 0);
    const bool mk1 = (mrow[kg + 16] == 0);
    float s0[8], s1[8];
#pragma unroll
    for (int v = 0; v < 8; ++v) {
      s0[v] = mk0 ? MNEG : sc0[v];
      s1[v] = mk1 ? MNEG : sc1[v];
      float mx = fmaxf(s0[v], s1[v]);
      mx = fmaxf(mx, __shfl_xor(mx, 1));
      mx = fmaxf(mx, __shfl_xor(mx, 2));
      mx = fmaxf(mx, __shfl_xor(mx, 4));
      mx = fmaxf(mx, __shfl_xor(mx, 8));
      if (l16 == 0) redbuf[wn][wm * 16 + half8 + v] = mx;
    }
    __syncthreads();
    if (tid < 64) {
      const float tm = fmaxf(redbuf[0][tid], redbuf[1][tid]);
      const float mo = row_m[tid];
      const float mn = fmaxf(mo, tm);
      row_m[tid] = mn;
      row_inv[tid] = __expf(mo - mn);        // sum correction factor (temp)
    }
    __syncthreads();
#pragma unroll
    for (int v = 0; v < 8; ++v) {
      const int row = wm * 16 + half8 + v;
      const float mn = row_m[row];
      float e = __expf(s0[v] - mn) + __expf(s1[v] - mn);
      e += __shfl_xor(e, 1);
      e += __shfl_xor(e, 2);
      e += __shfl_xor(e, 4);
      e += __shfl_xor(e, 8);
      if (l16 == 0) redbuf[wn][row] = e;
    }
    __syncthreads();
    if (tid < 64)
      row_l[tid] = row_l[tid] * row_inv[tid] + redbuf[0][tid] + redbuf[1][tid];
    __syncthreads();
  }

  if (wave == 0)
    tdm_load_tile64(lds_off(Ks[0]), kb0, 512);       // restart K stream
  if (tid < 64) row_inv[tid] = (row_l[tid] > 0.0f) ? (1.0f / row_l[tid]) : 0.0f;
  __syncthreads();

  // ---------------- pass 2: probabilities + O = P @ V ----------------
  v8f o0 = {}, o1 = {};
  for (int kt = 0; kt < 64; ++kt) {
    if (wave == 0) {
      if (kt + 1 < 64) {
        tdm_load_tile64(lds_off(Ks[(kt + 1) & 1]),
                        kb0 + (size_t)(kt + 1) * 64 * 512, 512);
        tdm_wait(1);
      } else {
        tdm_wait(0);
      }
    }
    // V tile, transposed to [d][key] for contiguous B-fragments
    {
      const us8* src = (const us8*)(vb0 + (size_t)(kt * 64 + lr) * 512 + lc);
      us8 t0 = src[0], t1 = src[1];
#pragma unroll
      for (int i = 0; i < 8; ++i) {
        Vt[(lc + i) * 72 + lr]     = t0[i];
        Vt[(lc + 8 + i) * 72 + lr] = t1[i];
      }
    }
    if (kt + 1 < 64)
      __builtin_prefetch(vb0 + (size_t)(kt + 1) * 64 * 512 + lr * 512 + lc, 0, 1);
    __syncthreads();
    const unsigned short* Kc = Ks[kt & 1];

    v8f sc0 = {}, sc1 = {};
#pragma unroll
    for (int ks = 0; ks < 64; ks += 32) {
      v16us a  = load_a(Qs, wm * 16, ks, lane);
      v16us b0 = load_b(Kc, wn * 32, ks, lane);
      v16us b1 = load_b(Kc, wn * 32 + 16, ks, lane);
      sc0 = wmma_bf16(a, b0, sc0);
      sc1 = wmma_bf16(a, b1, sc1);
    }
    const int kg = kt * 64 + wn * 32 + l16;
    const bool mk0 = (mrow[kg] == 0);
    const bool mk1 = (mrow[kg + 16] == 0);
#pragma unroll
    for (int v = 0; v < 8; ++v) {
      const int row = wm * 16 + half8 + v;
      const float mn = row_m[row];
      const float inv = row_inv[row];
      const float p0 = __expf((mk0 ? MNEG : sc0[v]) - mn) * inv;
      const float p1 = __expf((mk1 ? MNEG : sc1[v]) - mn) * inv;
      const size_t qg = (size_t)qt * 64 + row;
      const size_t ab = ((((size_t)n * 8 + h) * 4096 + qg) * 4096) + kg;
      att[ab]      = p0;
      att[ab + 16] = p1;
      Ps[row * 72 + wn * 32 + l16]      = f2bf(p0);
      Ps[row * 72 + wn * 32 + 16 + l16] = f2bf(p1);
    }
    __syncthreads();
#pragma unroll
    for (int ks = 0; ks < 64; ks += 32) {
      v16us a  = load_a(Ps, wm * 16, ks, lane);
      v16us b0 = load_b(Vt, wn * 32, ks, lane);
      v16us b1 = load_b(Vt, wn * 32 + 16, ks, lane);
      o0 = wmma_bf16(a, b0, o0);
      o1 = wmma_bf16(a, b1, o1);
    }
    __syncthreads();
  }
#pragma unroll
  for (int v = 0; v < 8; ++v) {
    const int row = wm * 16 + half8 + v;
    const size_t ob = (qrow0 + row) * 512 + h * 64 + wn * 32 + l16;
    Op[ob]      = f2bf(o0[v]);
    Op[ob + 16] = f2bf(o1[v]);
  }
}

// ---------------------------------------------------------------------------
// Kernel 4: out_f32[8192,512] = O_bf16[8192,512] @ Wt_o + b_o
// ---------------------------------------------------------------------------
__global__ __launch_bounds__(256) void out_gemm(
    const unsigned short* __restrict__ A, const unsigned short* __restrict__ Wt,
    const float* __restrict__ bias, float* __restrict__ Y) {
  __shared__ alignas(16) unsigned short As[64 * 72];
  __shared__ alignas(16) unsigned short Bt[64 * 72];
  const int tid = threadIdx.x, lane = tid & 31, wave = tid >> 5;
  const int wm = wave & 3, wn = wave >> 2;
  const int row0 = blockIdx.x * 64, col0 = blockIdx.y * 64;
  const int lr = tid >> 2, lc = (tid & 3) * 16;
  v8f acc0 = {}, acc1 = {};
  for (int kt = 0; kt < 512; kt += 64) {
    {
      const us8* src = (const us8*)(A + (size_t)(row0 + lr) * 512 + kt + lc);
      us8* dst = (us8*)(As + lr * 72 + lc);
      dst[0] = src[0]; dst[1] = src[1];
    }
    {
      const unsigned short* src = Wt + (size_t)(kt + lr) * 512 + col0 + lc;
#pragma unroll
      for (int i = 0; i < 16; ++i) Bt[(lc + i) * 72 + lr] = src[i];
    }
    __syncthreads();
#pragma unroll
    for (int ks = 0; ks < 64; ks += 32) {
      v16us a  = load_a(As, wm * 16, ks, lane);
      v16us b0 = load_b(Bt, wn * 32, ks, lane);
      v16us b1 = load_b(Bt, wn * 32 + 16, ks, lane);
      acc0 = wmma_bf16(a, b0, acc0);
      acc1 = wmma_bf16(a, b1, acc1);
    }
    __syncthreads();
  }
  const int half8 = (lane >> 4) * 8, l16 = lane & 15;
#pragma unroll
  for (int v = 0; v < 8; ++v) {
    const int r = row0 + wm * 16 + half8 + v;
    const int c = col0 + wn * 32 + l16;
    Y[(size_t)r * 512 + c]      = acc0[v] + bias[c];
    Y[(size_t)r * 512 + c + 16] = acc1[v] + bias[c + 16];
  }
}

// ---------------------------------------------------------------------------
extern "C" void kernel_launch(void* const* d_in, const int* in_sizes, int n_in,
                              void* d_out, int out_size, void* d_ws, size_t ws_size,
                              hipStream_t stream) {
  (void)in_sizes; (void)n_in; (void)out_size; (void)ws_size;
  const float* values  = (const float*)d_in[0];
  const float* keys    = (const float*)d_in[1];
  const float* queries = (const float*)d_in[2];
  const int*   mask    = (const int*)d_in[3];
  const float* W_v     = (const float*)d_in[4];
  const float* W_k     = (const float*)d_in[5];
  const float* W_q     = (const float*)d_in[6];
  const float* W_o     = (const float*)d_in[7];
  const float* b_o     = (const float*)d_in[8];

  float* out = (float*)d_out;                       // [2,4096,512]
  float* att = out + (size_t)2 * 4096 * 512;        // [2,8,4096,4096]

  unsigned short* Wt   = (unsigned short*)d_ws;     // 4 x [512,512] bf16
  unsigned short* Wt_q = Wt;
  unsigned short* Wt_k = Wt + 262144;
  unsigned short* Wt_v = Wt + 2 * 262144;
  unsigned short* Wt_o = Wt + 3 * 262144;
  unsigned short* Qp   = Wt + 4 * 262144;           // [8192,512] bf16 each
  unsigned short* Kp   = Qp + (size_t)8192 * 512;
  unsigned short* Vp   = Kp + (size_t)8192 * 512;
  unsigned short* Opb  = Vp + (size_t)8192 * 512;

  wt_convert<<<dim3(1024, 4), 256, 0, stream>>>(W_v, W_k, W_q, W_o, Wt);
  proj_gemm<<<dim3(128, 8), 256, 0, stream>>>(queries, Wt_q, Qp);
  proj_gemm<<<dim3(128, 8), 256, 0, stream>>>(keys,    Wt_k, Kp);
  proj_gemm<<<dim3(128, 8), 256, 0, stream>>>(values,  Wt_v, Vp);
  attn_kernel<<<dim3(64, 8, 2), 256, 0, stream>>>(Qp, Kp, Vp, mask, Opb, att);
  out_gemm<<<dim3(128, 8), 256, 0, stream>>>(Opb, Wt_o, b_o, out);
}
